// NALULayer_46033459478724
// MI455X (gfx1250) — matrix operand order repaired and checked by reading, not compiled, not run on hardware
//
#include <hip/hip_runtime.h>
#include <hip/hip_bf16.h>
#include <math.h>

#define DIM 128
#define NROWS 32768
#define EPS 1e-7f
#define OMEGA 20.0f

typedef __attribute__((ext_vector_type(16))) _Float16 v16h;
typedef __attribute__((ext_vector_type(8)))  float    v8f;

// Workspace layout:
//   [0, 98304)       : 3 B-matrices as WMMA B-fragments (f16), layout
//                      [mat][coltile(8)][ktile(4)][lane(32)][h(16)]  (32B per lane entry)
//   [98304, 98816)   : g[128] = tanh(G) (f32)
#define BFRAG_HALFS (3 * 8 * 4 * 32 * 16)   // 49152 halfs = 96 KB
#define G_OFFSET_BYTES 98304

// ---------------------------------------------------------------------------
// Prep: build B matrices in fragment layout.
// B-fragment layout used here (contiguous k order, same as A): lane L holds column n = L%16;
// halves h=0..15 hold K = (L/16)*16 + h  (lanes 0-15: K 0..15, lanes 16-31: K 16..31).
// 3072 threads total: t = ((mat*8 + ct)*4 + kt)*32 + lane
// ---------------------------------------------------------------------------
__global__ void __launch_bounds__(256)
nalu_prep(const float* __restrict__ W_m, const float* __restrict__ M_m,
          const float* __restrict__ G, _Float16* __restrict__ bfrag,
          float* __restrict__ g) {
  int t = blockIdx.x * blockDim.x + threadIdx.x;      // 0..3071
  int mat  = t >> 10;
  int rem  = t & 1023;
  int ct   = rem >> 7;
  int rem2 = rem & 127;
  int kt   = rem2 >> 5;
  int lane = rem2 & 31;
  int n     = ct * 16 + (lane & 15);
  int kbase = kt * 32 + (lane >> 4) * 16;

  v16h vv;
  #pragma unroll
  for (int h = 0; h < 16; ++h) {
    int k = kbase + h;
    float a;
    if (k == n) {
      a = 1.0f;
    } else {
      float w = W_m[k * DIM + n];
      float m = M_m[k * DIM + n];
      a = tanhf(w) * (1.0f / (1.0f + __expf(-m)));
    }
    float v;
    if (mat == 0) {
      v = a;                                   // weight_mul
    } else {
      float tt = 1.0f - 2.0f * fabsf(a);       // factor for sign = -1
      if (mat == 1) v = __logf(fmaxf(fabsf(tt), 1e-38f));  // log-magnitude
      else          v = (tt < 0.0f) ? 1.0f : 0.0f;         // negativity bit
    }
    vv[h] = (_Float16)v;
  }
  *(v16h*)(bfrag + (size_t)t * 16) = vv;

  if (t < DIM) g[t] = tanhf(G[t]);
}

// ---------------------------------------------------------------------------
// Main: 8 waves/block, each wave computes a 16-row x 128-col output tile via
// three fused WMMA GEMM chains + transcendental epilogue.
// ---------------------------------------------------------------------------
__global__ void __launch_bounds__(256)
nalu_main(const float* __restrict__ x, const _Float16* __restrict__ bfrag,
          const float* __restrict__ g, float* __restrict__ out) {
  extern __shared__ char smem[];                 // 96 KB of B fragments
  int tid = threadIdx.x;

  // Stage all B fragments into LDS (96KB / 256 threads = 24 x 16B each).
  {
    const uint4* src = (const uint4*)bfrag;
    uint4*       dst = (uint4*)smem;
    #pragma unroll
    for (int i = 0; i < 24; ++i)
      dst[tid + i * 256] = src[tid + i * 256];
  }
  __syncthreads();

  int wave = tid >> 5;
  int lane = tid & 31;
  int Mrow = lane & 15;       // row within tile (A/C layout)
  int grp  = lane >> 4;       // lane half-group
  int row0 = (blockIdx.x * 8 + wave) * 16;

  // Build A fragments: alx = log(max(|x|,eps)), aind = (x<0) indicator.
  // A layout used here (contiguous k, same as B): lane group g, halves 0..15 -> K = kt*32 + g*16 + h.
  const float* xrow = x + (size_t)(row0 + Mrow) * DIM;
  v16h alx[4], aind[4];
  #pragma unroll
  for (int kt = 0; kt < 4; ++kt) {
    int kb0 = kt * 32 + grp * 16;
    float4 f0 = *(const float4*)(xrow + kb0);
    float4 f1 = *(const float4*)(xrow + kb0 + 4);
    float4 f2 = *(const float4*)(xrow + kb0 + 8);
    float4 f3 = *(const float4*)(xrow + kb0 + 12);
    float xs[16] = {f0.x, f0.y, f0.z, f0.w, f1.x, f1.y, f1.z, f1.w,
                    f2.x, f2.y, f2.z, f2.w, f3.x, f3.y, f3.z, f3.w};
    #pragma unroll
    for (int h = 0; h < 16; ++h) {
      float xv = xs[h];
      alx[kt][h]  = (_Float16)__logf(fmaxf(fabsf(xv), EPS));
      aind[kt][h] = (_Float16)((xv < 0.0f) ? 1.0f : 0.0f);
    }
  }

  const v16h* blds = (const v16h*)smem;
  #pragma unroll 1
  for (int ct = 0; ct < 8; ++ct) {
    v8f c_ls = {}, c_lm = {}, c_pb = {};
    #pragma unroll
    for (int kt = 0; kt < 4; ++kt) {
      v16h ba = blds[((0 * 8 + ct) * 4 + kt) * 32 + lane];  // weight_mul
      v16h bL = blds[((1 * 8 + ct) * 4 + kt) * 32 + lane];  // log|1-2|w||
      v16h bP = blds[((2 * 8 + ct) * 4 + kt) * 32 + lane];  // neg bit
      c_ls = __builtin_amdgcn_wmma_f32_16x16x32_f16(false, alx[kt],  false, ba,
                                                    (short)0, c_ls, false, false);
      c_lm = __builtin_amdgcn_wmma_f32_16x16x32_f16(false, aind[kt], false, bL,
                                                    (short)0, c_lm, false, false);
      c_pb = __builtin_amdgcn_wmma_f32_16x16x32_f16(false, aind[kt], false, bP,
                                                    (short)0, c_pb, false, false);
    }

    // Epilogue. C/D layout: VGPR r -> row = grp*8 + r, col = ct*16 + (lane&15).
    int col = ct * 16 + Mrow;
    float gc = g[col];
    #pragma unroll
    for (int r = 0; r < 8; ++r) {
      int row = row0 + grp * 8 + r;
      float xv  = x[(size_t)row * DIM + col];
      float mul = __expf(fminf(c_ls[r], OMEGA));   // exp(min(log_space, 20))
      float msv = __expf(c_lm[r]);                 // magnitude of sign product
      int   par = ((int)(c_pb[r] + 0.5f)) & 1;     // parity of negative factors
      float sgn = par ? -1.0f : 1.0f;
      out[(size_t)row * DIM + col] = xv + mul * msv * sgn * gc;
    }
  }
}

extern "C" void kernel_launch(void* const* d_in, const int* in_sizes, int n_in,
                              void* d_out, int out_size, void* d_ws, size_t ws_size,
                              hipStream_t stream) {
  const float* x   = (const float*)d_in[0];
  const float* W_m = (const float*)d_in[1];
  const float* M_m = (const float*)d_in[2];
  const float* G   = (const float*)d_in[3];
  float* out = (float*)d_out;

  _Float16* bfrag = (_Float16*)d_ws;
  float*    g     = (float*)((char*)d_ws + G_OFFSET_BYTES);

  // Allow 96KB dynamic LDS (WGP has 320KB). Idempotent; called every time so
  // kernel_launch stays stateless/deterministic (no static guards).
  hipFuncSetAttribute((const void*)nalu_main,
                      hipFuncAttributeMaxDynamicSharedMemorySize, 98304);

  nalu_prep<<<12, 256, 0, stream>>>(W_m, M_m, G, bfrag, g);
  nalu_main<<<NROWS / (16 * 8), 256, 98304, stream>>>(x, bfrag, g, out);
}
